// SlidingWindowAttention_20658792694669
// MI455X (gfx1250) — matrix operand order repaired and checked by reading, not compiled, not run on hardware
//
#include <hip/hip_runtime.h>
#include <hip/hip_bf16.h>

typedef __attribute__((ext_vector_type(16))) _Float16 v16h;
typedef __attribute__((ext_vector_type(8)))  float    v8f;

#define BATCH   2
#define SEQ     2048
#define NHEADS  16
#define HDIM    64
#define DMODEL  1024
#define WINDOW  1024

union V16H { v16h v; float4 f4[2]; };

// ---------------------------------------------------------------------------
// Optional CDNA5 async global->LDS copies (ASYNCcnt path), with fallback.
// Builtin signature (from probe diagnostic): param0 = int4 __device__* (AS1),
// param1 = int4 __shared__* (AS3), then two int immediates (offset, cpol).
// ---------------------------------------------------------------------------
#if __has_builtin(__builtin_amdgcn_global_load_async_to_lds_b128) && \
    __has_builtin(__builtin_amdgcn_s_wait_asynccnt)
#define ASYNC_LDS 1
typedef int v4i_ __attribute__((vector_size(16)));
typedef __attribute__((address_space(1))) v4i_ glb_v4i;
typedef __attribute__((address_space(3))) v4i_ lds_v4i;
__device__ __forceinline__ void async_b128(const void* g, void* l) {
    __builtin_amdgcn_global_load_async_to_lds_b128(
        (glb_v4i*)(unsigned long long)g,
        (lds_v4i*)(unsigned)(unsigned long long)l, 0, 0);
}
__device__ __forceinline__ void async_wait0() {
    __builtin_amdgcn_s_wait_asynccnt(0);
}
#endif

// ---------------------------------------------------------------------------
// Elementwise f32 -> f16 conversion (16 elems/thread, fully vectorized).
// ---------------------------------------------------------------------------
__global__ __launch_bounds__(256)
void cvt_kernel(const float* __restrict__ src, _Float16* __restrict__ dst)
{
    const size_t i = ((size_t)blockIdx.x * 256 + threadIdx.x) * 16;
    float t[16];
    *(float4*)&t[0]  = *(const float4*)(src + i);
    *(float4*)&t[4]  = *(const float4*)(src + i + 4);
    *(float4*)&t[8]  = *(const float4*)(src + i + 8);
    *(float4*)&t[12] = *(const float4*)(src + i + 12);
    V16H u;
#pragma unroll
    for (int j = 0; j < 16; ++j) u.v[j] = (_Float16)t[j];
    *(float4*)(dst + i)     = u.f4[0];
    *(float4*)(dst + i + 8) = u.f4[1];
}

// ---------------------------------------------------------------------------
// Pure-f16 tiled GEMM, 128x128 block tile, 4 waves of 64x64 (16 WMMA/step),
// double-buffered LDS with async (or register) staging.
//   C[m][n] = sum_k A[m][k] * W[n][k] + bias[n],  M=4096, N=K=1024
// OUT_MODE: 0 -> f16 [B,H,S,D]   1 -> f16 [B,H,D,S]   2 -> f32 row-major
// ---------------------------------------------------------------------------
template<int OUT_MODE>
__global__ __launch_bounds__(128)
void gemm_kernel(const _Float16* __restrict__ A,
                 const _Float16* __restrict__ W,
                 const float* __restrict__ bias,
                 void* __restrict__ outp)
{
    const int N = DMODEL, K = DMODEL;

    __shared__ alignas(16) _Float16 As[2][128][32];   // [buf][m][k]
    __shared__ alignas(16) _Float16 Bs[2][128][32];   // [buf][n][k]

    const int m0   = blockIdx.x * 128;
    const int n0   = blockIdx.y * 128;
    const int tid  = threadIdx.x;        // 128 threads = 4 waves
    const int lane = tid & 31;
    const int wid  = tid >> 5;
    const int l16  = lane & 15;
    const int hh   = lane >> 4;
    const int wm   = (wid & 1) * 64;
    const int wn   = (wid >> 1) * 64;

    const _Float16* arow = A + (size_t)(m0 + tid) * K;   // this thread stages A row tid
    const _Float16* wrow = W + (size_t)(n0 + tid) * K;   // and W row tid (32 halves each)

    v8f acc[4][4];
#pragma unroll
    for (int mi = 0; mi < 4; ++mi)
#pragma unroll
        for (int ni = 0; ni < 4; ++ni)
#pragma unroll
            for (int v = 0; v < 8; ++v) acc[mi][ni][v] = 0.0f;

    // ---- prologue: stage k0 = 0 into buffer 0 ----
#ifdef ASYNC_LDS
    {
#pragma unroll
        for (int c = 0; c < 32; c += 8) {
            async_b128(arow + c, &As[0][tid][c]);
            async_b128(wrow + c, &Bs[0][tid][c]);
        }
        async_wait0();
    }
#else
    {
#pragma unroll
        for (int c = 0; c < 32; c += 8) {
            *(float4*)&As[0][tid][c] = *(const float4*)(arow + c);
            *(float4*)&Bs[0][tid][c] = *(const float4*)(wrow + c);
        }
    }
#endif
    __syncthreads();

    for (int k0 = 0; k0 < K; k0 += 32) {
        const int  cur  = (k0 >> 5) & 1;
        const bool more = (k0 + 32) < K;
        const int  kn   = k0 + 32;

#ifdef ASYNC_LDS
        if (more) {
#pragma unroll
            for (int c = 0; c < 32; c += 8) {
                async_b128(arow + kn + c, &As[cur ^ 1][tid][c]);
                async_b128(wrow + kn + c, &Bs[cur ^ 1][tid][c]);
            }
        }
#else
        float4 ar[4], br[4];
        if (more) {
#pragma unroll
            for (int c = 0; c < 4; ++c) {
                ar[c] = *(const float4*)(arow + kn + c * 8);
                br[c] = *(const float4*)(wrow + kn + c * 8);
            }
        }
#endif

        // ---- compute on current buffer: 64x64 per wave, 16 WMMAs ----
        v16h bfr[4];
#pragma unroll
        for (int ni = 0; ni < 4; ++ni) {
            V16H u;
            const _Float16* p = &Bs[cur][wn + ni * 16 + l16][16 * hh];
            u.f4[0] = *(const float4*)(p);
            u.f4[1] = *(const float4*)(p + 8);
            bfr[ni] = u.v;
        }
#pragma unroll
        for (int mi = 0; mi < 4; ++mi) {
            V16H u;
            const _Float16* p = &As[cur][wm + mi * 16 + l16][0];
            u.f4[0] = *(const float4*)(p + 16 * hh);
            u.f4[1] = *(const float4*)(p + 16 * hh + 8);
            const v16h afr = u.v;
#pragma unroll
            for (int ni = 0; ni < 4; ++ni)
                acc[mi][ni] = __builtin_amdgcn_wmma_f32_16x16x32_f16(
                    false, afr, false, bfr[ni], (short)0, acc[mi][ni], false, false);
        }

#ifdef ASYNC_LDS
        if (more) async_wait0();
#else
        if (more) {
#pragma unroll
            for (int c = 0; c < 4; ++c) {
                *(float4*)&As[cur ^ 1][tid][c * 8] = ar[c];
                *(float4*)&Bs[cur ^ 1][tid][c * 8] = br[c];
            }
        }
#endif
        __syncthreads();
    }

    // ---- store D ----
#pragma unroll
    for (int mi = 0; mi < 4; ++mi)
#pragma unroll
        for (int ni = 0; ni < 4; ++ni)
#pragma unroll
            for (int v = 0; v < 8; ++v) {
                const int m = m0 + wm + mi * 16 + v + 8 * hh;
                const int n = n0 + wn + ni * 16 + l16;
                const float val = acc[mi][ni][v] + bias[n];
                if (OUT_MODE == 2) {
                    ((float*)outp)[(size_t)m * N + n] = val;
                } else {
                    const int b = m >> 11;
                    const int s = m & (SEQ - 1);
                    const int h = n >> 6;
                    const int d = n & (HDIM - 1);
                    _Float16* o = (_Float16*)outp;
                    if (OUT_MODE == 0)
                        o[(((size_t)(b * NHEADS + h) * SEQ + s) << 6) + d] = (_Float16)val;
                    else
                        o[((size_t)(b * NHEADS + h) * HDIM + d) * SEQ + s] = (_Float16)val;
                }
            }
}

// ---------------------------------------------------------------------------
// Sliding-window attention, ALiBi, flash-style online softmax.
// 8 waves/block share double-buffered K/V chunks in LDS (staged once/block).
// Transposed-score trick: D-layout of probs == A-fragment layout for P*V.
// qh,kh: [B,H,S,64] f16;  vt: [B,H,64,S] f16;  oh: [B,S,1024] f16
// ---------------------------------------------------------------------------
__global__ __launch_bounds__(256)
void attn_kernel(const _Float16* __restrict__ qh,
                 const _Float16* __restrict__ kh,
                 const _Float16* __restrict__ vt,
                 _Float16* __restrict__ oh)
{
    __shared__ alignas(16) _Float16 Ks[2][32][64];   // [buf][c][d]
    __shared__ alignas(16) _Float16 Vs[2][64][32];   // [buf][d][c]

    const int tid  = threadIdx.x;
    const int wid  = tid >> 5;
    const int lane = tid & 31;
    const int l16  = lane & 15;
    const int hh   = lane >> 4;

    const int blk = blockIdx.x;
    const int qt  = blk & 15;
    const int bh  = blk >> 4;
    const int h   = bh & (NHEADS - 1);
    const int bb  = bh >> 4;

    const int Q0 = qt * 128;
    const int q0 = Q0 + wid * 16;
    const float slope = exp2f(-0.5f * (float)(h + 1));

    const _Float16* qbase = qh + (size_t)bh * SEQ * HDIM;
    const _Float16* kbase = kh + (size_t)bh * SEQ * HDIM;
    const _Float16* vbase = vt + (size_t)bh * HDIM * SEQ;

    const int krow = tid >> 3;               // 0..31
    const int kcol = (tid & 7) * 8;
    const int vrow = tid >> 2;               // 0..63
    const int vcol = (tid & 3) * 8;

    int clo = Q0 - (WINDOW - 1);
    if (clo < 0) clo = 0;
    clo &= ~31;
    const int nch = ((Q0 + 127 - clo) >> 5) + 1;

    const int cw_lo = (q0 - (WINDOW - 1)) - 31;
    const int cw_hi = q0 + 15;

    v16h qf[2];
#pragma unroll
    for (int t = 0; t < 2; ++t) {
        V16H u;
        const _Float16* p = qbase + (size_t)(q0 + l16) * HDIM + t * 32 + 16 * hh;
        u.f4[0] = *(const float4*)(p);
        u.f4[1] = *(const float4*)(p + 8);
        qf[t] = u.v;
    }

    v8f oacc[4];
#pragma unroll
    for (int dt = 0; dt < 4; ++dt)
#pragma unroll
        for (int v = 0; v < 8; ++v) oacc[dt][v] = 0.0f;

    float mrun = -1e30f;
    float lrun = 0.0f;
    const int rl = 8 * hh - l16 - q0;

    // ---- prologue: stage chunk 0 ----
#ifdef ASYNC_LDS
    {
        async_b128(kbase + (size_t)(clo + krow) * HDIM + kcol, &Ks[0][krow][kcol]);
        async_b128(vbase + (size_t)vrow * SEQ + clo + vcol,    &Vs[0][vrow][vcol]);
        async_wait0();
    }
#else
    {
        *(float4*)&Ks[0][krow][kcol] =
            *(const float4*)(kbase + (size_t)(clo + krow) * HDIM + kcol);
        *(float4*)&Vs[0][vrow][vcol] =
            *(const float4*)(vbase + (size_t)vrow * SEQ + clo + vcol);
    }
#endif
    __syncthreads();

    for (int i = 0; i < nch; ++i) {
        const int  c0   = clo + i * 32;
        const int  buf  = i & 1;
        const bool more = (i + 1) < nch;
        const int  cn   = c0 + 32;

#ifdef ASYNC_LDS
        if (more) {
            async_b128(kbase + (size_t)(cn + krow) * HDIM + kcol, &Ks[buf ^ 1][krow][kcol]);
            async_b128(vbase + (size_t)vrow * SEQ + cn + vcol,    &Vs[buf ^ 1][vrow][vcol]);
        }
#else
        float4 nk, nv;
        if (more) {
            nk = *(const float4*)(kbase + (size_t)(cn + krow) * HDIM + kcol);
            nv = *(const float4*)(vbase + (size_t)vrow * SEQ + cn + vcol);
        }
#endif

        const bool active = (c0 >= cw_lo) && (c0 <= cw_hi);   // wave-uniform
        if (active) {
            // ---- transposed scores St[c][r] = K[c][:] . q[r][:] ----
            float sc[2][8];
            const int   d0  = c0 + rl;
            const float fd0 = (float)d0;
#pragma unroll
            for (int t = 0; t < 2; ++t) {
                v8f s;
#pragma unroll
                for (int v = 0; v < 8; ++v) s[v] = 0.0f;
#pragma unroll
                for (int dd = 0; dd < 2; ++dd) {
                    V16H u;
                    const _Float16* p = &Ks[buf][t * 16 + l16][dd * 32];
                    u.f4[0] = *(const float4*)(p + 16 * hh);
                    u.f4[1] = *(const float4*)(p + 16 * hh + 8);
                    s = __builtin_amdgcn_wmma_f32_16x16x32_f16(
                            false, u.v, false, qf[dd], (short)0, s, false, false);
                }
#pragma unroll
                for (int v = 0; v < 8; ++v) {
                    const int  de    = d0 + 16 * t + v;
                    const bool valid = (unsigned)(de + (WINDOW - 1)) < (unsigned)WINDOW;
                    const float val = fmaf(slope, fd0 + (float)(16 * t + v), s[v] * 0.125f);
                    sc[t][v] = valid ? val : -1e30f;
                }
            }

            // ---- online softmax ----
            float mc = -1e30f;
#pragma unroll
            for (int t = 0; t < 2; ++t)
#pragma unroll
                for (int v = 0; v < 8; ++v) mc = fmaxf(mc, sc[t][v]);
            mc = fmaxf(mc, __shfl_xor(mc, 16, 32));
            const float mnew  = fmaxf(mrun, mc);
            const float scale = __expf(mrun - mnew);

            float ps = 0.0f;
            v16h pf;
#pragma unroll
            for (int t = 0; t < 2; ++t)
#pragma unroll
                for (int v = 0; v < 8; ++v) {
                    const float p = (sc[t][v] > -9e29f) ? __expf(sc[t][v] - mnew) : 0.0f;
                    ps += p;
                    pf[t * 8 + v] = (_Float16)p;
                }
            ps += __shfl_xor(ps, 16, 32);
            lrun = lrun * scale + ps;
            mrun = mnew;

#pragma unroll
            for (int v = 0; v < 8; ++v) {
                const float scv = __shfl(scale, v + 8 * hh, 32);
#pragma unroll
                for (int dt = 0; dt < 4; ++dt) oacc[dt][v] *= scv;
            }

            // ---- O += P * V ----
#pragma unroll
            for (int dt = 0; dt < 4; ++dt) {
                V16H u;
                const _Float16* p = &Vs[buf][dt * 16 + l16][0];
                u.f4[0] = *(const float4*)(p + 8 * hh);
                u.f4[1] = *(const float4*)(p + 16 + 8 * hh);
                oacc[dt] = __builtin_amdgcn_wmma_f32_16x16x32_f16(
                               false, pf, false, u.v, (short)0, oacc[dt], false, false);
            }
        }

#ifdef ASYNC_LDS
        if (more) async_wait0();
#else
        if (more) {
            *(float4*)&Ks[buf ^ 1][krow][kcol] = nk;
            *(float4*)&Vs[buf ^ 1][vrow][vcol] = nv;
        }
#endif
        __syncthreads();
    }

    // ---- finalize ----
    const float linv = 1.0f / lrun;
#pragma unroll
    for (int v = 0; v < 8; ++v) {
        const float lv = __shfl(linv, v + 8 * hh, 32);
        const int r = v + 8 * hh;
#pragma unroll
        for (int dt = 0; dt < 4; ++dt) {
            const size_t off = ((size_t)bb * SEQ + q0 + r) * DMODEL
                             + (size_t)h * HDIM + dt * 16 + l16;
            oh[off] = (_Float16)(oacc[dt][v] * lv);
        }
    }
}

// ---------------------------------------------------------------------------
extern "C" void kernel_launch(void* const* d_in, const int* in_sizes, int n_in,
                              void* d_out, int out_size, void* d_ws, size_t ws_size,
                              hipStream_t stream)
{
    const float* x  = (const float*)d_in[0];
    const float* wq = (const float*)d_in[1];
    const float* bq = (const float*)d_in[2];
    const float* wk = (const float*)d_in[3];
    const float* bk = (const float*)d_in[4];
    const float* wv = (const float*)d_in[5];
    const float* bv = (const float*)d_in[6];
    const float* wo = (const float*)d_in[7];
    const float* bo = (const float*)d_in[8];
    float* out = (float*)d_out;

    _Float16* ws = (_Float16*)d_ws;
    const size_t BUF  = (size_t)BATCH * SEQ * DMODEL;   // 4 Mi halves
    const size_t WBUF = (size_t)DMODEL * DMODEL;        // 1 Mi halves
    _Float16* qhb = ws;
    _Float16* khb = ws + BUF;
    _Float16* vth = ws + 2 * BUF;
    _Float16* oh  = ws + 3 * BUF;
    _Float16* xh  = ws + 4 * BUF;
    _Float16* wqh = ws + 5 * BUF;
    _Float16* wkh = wqh + WBUF;
    _Float16* wvh = wqh + 2 * WBUF;
    _Float16* woh = wqh + 3 * WBUF;

    // one-time f32 -> f16 conversion of activations and weights
    cvt_kernel<<<(int)(BUF  / (16 * 256)), 256, 0, stream>>>(x,  xh);
    cvt_kernel<<<(int)(WBUF / (16 * 256)), 256, 0, stream>>>(wq, wqh);
    cvt_kernel<<<(int)(WBUF / (16 * 256)), 256, 0, stream>>>(wk, wkh);
    cvt_kernel<<<(int)(WBUF / (16 * 256)), 256, 0, stream>>>(wv, wvh);
    cvt_kernel<<<(int)(WBUF / (16 * 256)), 256, 0, stream>>>(wo, woh);

    dim3 gg(32, 8);
    gemm_kernel<0><<<gg, 128, 0, stream>>>(xh, wqh, bq, qhb);
    gemm_kernel<0><<<gg, 128, 0, stream>>>(xh, wkh, bk, khb);
    gemm_kernel<1><<<gg, 128, 0, stream>>>(xh, wvh, bv, vth);

    attn_kernel<<<BATCH * NHEADS * (SEQ / 128), 256, 0, stream>>>(qhb, khb, vth, oh);

    gemm_kernel<2><<<gg, 128, 0, stream>>>(oh, woh, bo, out);
}
